// GrapHiC_35416300323765
// MI455X (gfx1250) — matrix operand order repaired and checked by reading, not compiled, not run on hardware
//
#include <hip/hip_runtime.h>
#include <hip/hip_bf16.h>

// ---------------------------------------------------------------------------
// GrapHiC graph-transformer forward, MI455X (gfx1250, wave32, WMMA).
//
// H=4, C=32, D=128, N=16384, E=524288, N_GRAPHS=16, N_PER=1024.
//
// Dense GEMMs: v_wmma_f32_16x16x32_f16 (f16 in, f32 accum).
// Edge stage is memory-bound; rank-1 edge embedding e = w*We + be is folded:
//   alpha_raw = (q.k + w*(q.We) + (q.be)) / sqrt(C)
//   out       = scatter(alpha*v) + Sw*We + S1*be
// so only q/k/v rows are gathered and only alpha*v is scattered.
//
// q/k/v/skip outputs live in ONE contiguous f16 buffer [4][N][128] so the
// WMMA store path is branchless (EXEC stays all-ones, no divergent stores).
// ---------------------------------------------------------------------------

#define NN      16384
#define EE      524288
#define HH      4
#define CC      32
#define DD      128
#define NGRAPH  16
#define NPER    1024
#define RS_C    0.17677669529663687f   // 1/sqrt(32)

typedef __attribute__((ext_vector_type(16))) _Float16 v16h;
typedef __attribute__((ext_vector_type(8)))  float    v8f;

union Frag16 { v16h v; _Float16 h[16]; float4 f4[2]; };
union H8     { float4 f; _Float16 h[8]; };

// Order-preserving f32 -> u32 key for atomic max (handles negatives).
__device__ __forceinline__ unsigned fkey(float x) {
  unsigned u = __float_as_uint(x);
  return (u & 0x80000000u) ? ~u : (u | 0x80000000u);
}
__device__ __forceinline__ float fdecode(unsigned k) {
  return (k & 0x80000000u) ? __uint_as_float(k ^ 0x80000000u)
                           : __uint_as_float(~k);
}

// ---------------------------------------------------------------------------
// Weight packing: WallT[512][32] f16 = [Wq|Wk|Wv|Wskip] columns, K padded to
// 32 (column-major => contiguous per-lane B-fragment loads). ball[512] = the
// four biases. WlinT[32][128] f16 = Wlin^T.
// ---------------------------------------------------------------------------
__global__ void k_pack(const float* __restrict__ Wq, const float* __restrict__ Wk,
                       const float* __restrict__ Wv, const float* __restrict__ Ws,
                       const float* __restrict__ bq, const float* __restrict__ bk,
                       const float* __restrict__ bv, const float* __restrict__ bs,
                       const float* __restrict__ Wlin, int in_dim,
                       _Float16* __restrict__ WallT, float* __restrict__ ball,
                       _Float16* __restrict__ WlinT)
{
  int t = blockIdx.x * blockDim.x + threadIdx.x;
  int stride = gridDim.x * blockDim.x;
  for (int idx = t; idx < 512 * 32; idx += stride) {
    int n = idx >> 5, k = idx & 31;
    int m = n >> 7, cc = n & 127;
    const float* W = (m == 0) ? Wq : (m == 1) ? Wk : (m == 2) ? Wv : Ws;
    float v = (k < in_dim) ? W[k * DD + cc] : 0.0f;
    WallT[n * 32 + k] = (_Float16)v;
  }
  for (int n = t; n < 512; n += stride) {
    int m = n >> 7, cc = n & 127;
    const float* b = (m == 0) ? bq : (m == 1) ? bk : (m == 2) ? bv : bs;
    ball[n] = b[cc];
  }
  for (int idx = t; idx < 32 * 128; idx += stride) {
    int n = idx >> 7, k = idx & 127;
    WlinT[n * 128 + k] = (_Float16)Wlin[k * 32 + n];
  }
}

// x (f32, [N,in_dim]) -> x16 (f16, [N,32], K zero-padded)
__global__ void k_cast_x(const float* __restrict__ x, int in_dim,
                         _Float16* __restrict__ x16)
{
  int t = blockIdx.x * blockDim.x + threadIdx.x;
  if (t >= NN * 32) return;
  int n = t >> 5, k = t & 31;
  x16[t] = (k < in_dim) ? (_Float16)x[n * in_dim + k] : (_Float16)0.0f;
}

// ---------------------------------------------------------------------------
// Fused q/k/v/skip projections: [N,32pad] @ [32,512] via WMMA 16x16x32.
// Grid: N/16 node tiles; 8 waves/block; wave owns col tiles wave*4..wave*4+3,
// which all fall in ONE of the four output matrices (m = wave>>1), so the
// store base pointer is wave-uniform and the store loop is branchless.
// Output: qkvs16 = [m=0:q | m=1:k | m=2:v | m=3:skip][N][128], f16.
// ---------------------------------------------------------------------------
__global__ __launch_bounds__(256) void k_qkvs(
    const _Float16* __restrict__ x16, const _Float16* __restrict__ WallT,
    const float* __restrict__ ball, _Float16* __restrict__ qkvs16)
{
  const int tile = blockIdx.x;
  const int wave = threadIdx.x >> 5;
  const int lane = threadIdx.x & 31;
  const int lo = lane & 15, hi = lane >> 4;

  __builtin_amdgcn_s_wait_tensorcnt(0);   // CDNA5 tensor counter (no-op fence)

  Frag16 a;
  const _Float16* rowp = x16 + (size_t)(tile * 16 + lo) * 32;
  a.f4[0] = *(const float4*)(rowp + 8 * hi);        // halves 0..7 : K = 8*hi..
  a.f4[1] = *(const float4*)(rowp + 16 + 8 * hi);   // halves 8..15: K = 16+8*hi..

  const int m = wave >> 1;                // which output matrix (uniform)
  _Float16* __restrict__ mbase = qkvs16 + (size_t)m * NN * DD;

#pragma unroll
  for (int i = 0; i < 4; ++i) {
    int ct = wave * 4 + i;                // col tile 0..31
    int n  = ct * 16 + lo;                // global output col 0..511
    int cc = ((ct & 7) * 16) + lo;        // col within this matrix, 0..127
    Frag16 b;
    const _Float16* colp = WallT + (size_t)n * 32 + 8 * hi;
    b.f4[0] = *(const float4*)(colp);
    b.f4[1] = *(const float4*)(colp + 16);

    v8f c = {};
    c = __builtin_amdgcn_wmma_f32_16x16x32_f16(false, a.v, false, b.v,
                                               (short)0, c, false, false);
    float bias = ball[n];
    _Float16* dp = mbase + (size_t)(tile * 16 + 8 * hi) * DD + cc;
#pragma unroll
    for (int r = 0; r < 8; ++r) {         // D: M = r + 8*hi, N = lo
      dp[(size_t)r * DD] = (_Float16)(c[r] + bias);
    }
  }
}

// Per (node, head): qWe = q.We, qbe = q.be; reset softmax/scatter accumulators.
__global__ void k_node_init(const _Float16* __restrict__ q16,
                            const float* __restrict__ We, const float* __restrict__ be,
                            float* __restrict__ qWe, float* __restrict__ qbe,
                            unsigned* __restrict__ amaxU, float* __restrict__ denom,
                            float* __restrict__ S1, float* __restrict__ Sw)
{
  int t = blockIdx.x * blockDim.x + threadIdx.x;
  if (t >= NN * HH) return;
  int n = t >> 2, h = t & 3;
  const _Float16* qp = q16 + (size_t)n * DD + h * CC;
  float a = 0.0f, b = 0.0f;
#pragma unroll 8
  for (int c = 0; c < CC; ++c) {
    float q = (float)qp[c];
    a += q * We[h * CC + c];
    b += q * be[h * CC + c];
  }
  qWe[t] = a; qbe[t] = b;
  amaxU[t] = 0x007FFFFFu;   // fkey(-inf)
  denom[t] = 0.0f; S1[t] = 0.0f; Sw[t] = 0.0f;
}

__global__ void k_zero(float* __restrict__ p, int count)
{
  int t = blockIdx.x * blockDim.x + threadIdx.x;
  if (t < count) p[t] = 0.0f;
}

// Edge alpha + running segment max. One thread per (edge, head).
__global__ void k_alpha(const int* __restrict__ srcv, const int* __restrict__ dstv,
                        const float* __restrict__ ea,
                        const _Float16* __restrict__ q16, const _Float16* __restrict__ k16,
                        const _Float16* __restrict__ v16,
                        const float* __restrict__ qWe, const float* __restrict__ qbe,
                        float* __restrict__ alpha, unsigned* __restrict__ amaxU)
{
  int t = blockIdx.x * blockDim.x + threadIdx.x;
  if (t >= EE * HH) return;
  int e = t >> 2, h = t & 3;
  int s = srcv[e], d = dstv[e];
  const _Float16* qp = q16 + (size_t)d * DD + h * CC;
  const _Float16* kp = k16 + (size_t)s * DD + h * CC;
  __builtin_prefetch((const void*)(v16 + (size_t)s * DD + h * CC), 0, 1);
  float acc = 0.0f;
#pragma unroll
  for (int ch = 0; ch < 4; ++ch) {
    H8 qa, ka;
    qa.f = *(const float4*)(qp + ch * 8);
    ka.f = *(const float4*)(kp + ch * 8);
#pragma unroll
    for (int j = 0; j < 8; ++j) acc += (float)qa.h[j] * (float)ka.h[j];
  }
  float w  = ea[e];
  float ar = (acc + w * qWe[d * HH + h] + qbe[d * HH + h]) * RS_C;
  alpha[t] = ar;
  atomicMax(&amaxU[d * HH + h], fkey(ar));
}

// exp(alpha - max) and segment denom.
__global__ void k_expsum(const int* __restrict__ dstv,
                         const unsigned* __restrict__ amaxU,
                         float* __restrict__ alpha, float* __restrict__ denom)
{
  int t = blockIdx.x * blockDim.x + threadIdx.x;
  if (t >= EE * HH) return;
  int e = t >> 2, h = t & 3;
  int d = dstv[e];
  float ex = expf(alpha[t] - fdecode(amaxU[d * HH + h]));
  alpha[t] = ex;
  atomicAdd(&denom[d * HH + h], ex);
}

// Normalize alpha, accumulate S1/Sw, scatter alpha * v[src].
__global__ void k_scatter(const int* __restrict__ srcv, const int* __restrict__ dstv,
                          const float* __restrict__ ea,
                          const float* __restrict__ alpha, const float* __restrict__ denom,
                          const _Float16* __restrict__ v16,
                          float* __restrict__ S1, float* __restrict__ Sw,
                          float* __restrict__ out_acc)
{
  int t = blockIdx.x * blockDim.x + threadIdx.x;
  if (t >= EE * HH) return;
  int e = t >> 2, h = t & 3;
  int s = srcv[e], d = dstv[e];
  float a = alpha[t] / (denom[d * HH + h] + 1e-16f);
  atomicAdd(&S1[d * HH + h], a);
  atomicAdd(&Sw[d * HH + h], a * ea[e]);
  const _Float16* vp = v16 + (size_t)s * DD + h * CC;
  float* op = out_acc + (size_t)d * DD + h * CC;
#pragma unroll
  for (int ch = 0; ch < 4; ++ch) {
    H8 va; va.f = *(const float4*)(vp + ch * 8);
#pragma unroll
    for (int j = 0; j < 8; ++j) atomicAdd(&op[ch * 8 + j], a * (float)va.h[j]);
  }
}

// One wave per node: reconstruct out (+= Sw*We + S1*be), beta gate, blend,
// write f16 for the Wlin WMMA. 8 nodes per 256-thread block.
__global__ __launch_bounds__(256) void k_gate(
    const float* __restrict__ out_acc, const _Float16* __restrict__ xr16,
    const float* __restrict__ S1, const float* __restrict__ Sw,
    const float* __restrict__ We, const float* __restrict__ be,
    const float* __restrict__ Wbeta, const float* __restrict__ bbeta,
    _Float16* __restrict__ outf16)
{
  int node = blockIdx.x * 8 + (threadIdx.x >> 5);
  int lane = threadIdx.x & 31;
  int c0 = lane * 4;
  int h  = c0 >> 5;            // all 4 channels within one head
  float S1n = S1[node * HH + h], Swn = Sw[node * HH + h];
  float o[4], xv[4], part = 0.0f;
#pragma unroll
  for (int j = 0; j < 4; ++j) {
    int c = c0 + j;
    o[j]  = out_acc[(size_t)node * DD + c] + Swn * We[c] + S1n * be[c];
    xv[j] = (float)xr16[(size_t)node * DD + c];
    part += o[j] * Wbeta[c] + xv[j] * Wbeta[DD + c] + (o[j] - xv[j]) * Wbeta[2 * DD + c];
  }
  for (int off = 16; off > 0; off >>= 1) part += __shfl_xor(part, off, 32);
  float bg = 1.0f / (1.0f + expf(-(part + bbeta[0])));
#pragma unroll
  for (int j = 0; j < 4; ++j) {
    float f = bg * xv[j] + (1.0f - bg) * o[j];
    outf16[(size_t)node * DD + c0 + j] = (_Float16)f;
  }
}

// out_final [N,128] @ Wlin [128,32] + blin, ReLU -> hbuf f32. WMMA, 4 K-steps.
__global__ __launch_bounds__(64) void k_lin(
    const _Float16* __restrict__ outf16, const _Float16* __restrict__ WlinT,
    const float* __restrict__ blin, float* __restrict__ hout)
{
  const int tile = blockIdx.x;
  const int wave = threadIdx.x >> 5;     // 2 waves: cols 0-15, 16-31
  const int lane = threadIdx.x & 31;
  const int lo = lane & 15, hi = lane >> 4;
  v8f c = {};
#pragma unroll
  for (int kk = 0; kk < 4; ++kk) {
    Frag16 a, b;
    const _Float16* rowp = outf16 + (size_t)(tile * 16 + lo) * DD + kk * 32;
    a.f4[0] = *(const float4*)(rowp + 8 * hi);
    a.f4[1] = *(const float4*)(rowp + 16 + 8 * hi);
    int n = wave * 16 + lo;
    const _Float16* colp = WlinT + (size_t)n * DD + kk * 32 + 8 * hi;
    b.f4[0] = *(const float4*)(colp);
    b.f4[1] = *(const float4*)(colp + 16);
    c = __builtin_amdgcn_wmma_f32_16x16x32_f16(false, a.v, false, b.v,
                                               (short)0, c, false, false);
  }
  int n = wave * 16 + lo;
  float bias = blin[n];
#pragma unroll
  for (int r = 0; r < 8; ++r) {
    int node = tile * 16 + r + 8 * hi;
    float v = c[r] + bias;
    hout[(size_t)node * CC + n] = v > 0.0f ? v : 0.0f;
  }
}

// GraphNorm: per-(graph, channel) mean.
__global__ void k_gn_mean(const float* __restrict__ hbuf, float* __restrict__ gmean)
{
  int t = blockIdx.x * blockDim.x + threadIdx.x;
  if (t >= NGRAPH * CC) return;
  int g = t >> 5, c = t & 31;
  const float* p = hbuf + (size_t)g * NPER * CC + c;
  float s = 0.0f;
  for (int i = 0; i < NPER; ++i) s += p[(size_t)i * CC];
  gmean[t] = s / (float)NPER;
}

// GraphNorm: shift by mean*mean_scale, variance, scale + bias -> outp.
__global__ void k_gn_norm(const float* __restrict__ hbuf, const float* __restrict__ gmean,
                          const float* __restrict__ mean_scale, const float* __restrict__ gamma,
                          const float* __restrict__ beta_gn, float* __restrict__ outp)
{
  int t = blockIdx.x * blockDim.x + threadIdx.x;
  if (t >= NGRAPH * CC) return;
  int g = t >> 5, c = t & 31;
  float m = gmean[t] * mean_scale[c];
  const float* p = hbuf + (size_t)g * NPER * CC + c;
  float vs = 0.0f;
  for (int i = 0; i < NPER; ++i) { float hs = p[(size_t)i * CC] - m; vs += hs * hs; }
  float scale = gamma[c] * rsqrtf(vs / (float)NPER + 1e-5f);
  float* q = outp + (size_t)g * NPER * CC + c;
  for (int i = 0; i < NPER; ++i) q[(size_t)i * CC] = scale * (p[(size_t)i * CC] - m) + beta_gn[c];
}

// ---------------------------------------------------------------------------
// Host side
// ---------------------------------------------------------------------------

// params flatten as JAX pytree (dict keys sorted): b0 then b1, keys:
// Wbeta, We, Wk, Wlin, Wq, Wskip, Wv, bbeta, be, beta_gn, bk, blin, bq,
// bskip, bv, gamma, mean_scale  (17 per block).
struct BP {
  const float *Wbeta, *We, *Wk, *Wlin, *Wq, *Wskip, *Wv;
  const float *bbeta, *be, *beta_gn, *bk, *blin, *bq, *bskip, *bv;
  const float *gamma, *mean_scale;
};
static BP get_bp(void* const* d_in, int base) {
  BP p;
  p.Wbeta     = (const float*)d_in[base + 0];
  p.We        = (const float*)d_in[base + 1];
  p.Wk        = (const float*)d_in[base + 2];
  p.Wlin      = (const float*)d_in[base + 3];
  p.Wq        = (const float*)d_in[base + 4];
  p.Wskip     = (const float*)d_in[base + 5];
  p.Wv        = (const float*)d_in[base + 6];
  p.bbeta     = (const float*)d_in[base + 7];
  p.be        = (const float*)d_in[base + 8];
  p.beta_gn   = (const float*)d_in[base + 9];
  p.bk        = (const float*)d_in[base + 10];
  p.blin      = (const float*)d_in[base + 11];
  p.bq        = (const float*)d_in[base + 12];
  p.bskip     = (const float*)d_in[base + 13];
  p.bv        = (const float*)d_in[base + 14];
  p.gamma     = (const float*)d_in[base + 15];
  p.mean_scale= (const float*)d_in[base + 16];
  return p;
}

struct WSBuf {
  _Float16 *qkvs16, *outf16, *x16, *WallT, *WlinT;
  float *out_acc, *alpha, *qWe, *qbe, *denom, *S1, *Sw, *ball, *gmean, *h0, *hbuf;
  unsigned *amaxU;
};
static char* carve(char*& p, size_t bytes) {
  char* r = p;
  p += (bytes + 255) & ~(size_t)255;
  return r;
}

static void run_block(hipStream_t stream, int in_dim, const float* x,
                      const BP& P, const WSBuf& w, float* outp,
                      const int* srcv, const int* dstv, const float* ea)
{
  const _Float16* q16  = w.qkvs16 + (size_t)0 * NN * DD;
  const _Float16* k16  = w.qkvs16 + (size_t)1 * NN * DD;
  const _Float16* v16  = w.qkvs16 + (size_t)2 * NN * DD;
  const _Float16* xr16 = w.qkvs16 + (size_t)3 * NN * DD;

  k_pack<<<16, 256, 0, stream>>>(P.Wq, P.Wk, P.Wv, P.Wskip, P.bq, P.bk, P.bv, P.bskip,
                                 P.Wlin, in_dim, w.WallT, w.ball, w.WlinT);
  k_cast_x<<<(NN * 32) / 256, 256, 0, stream>>>(x, in_dim, w.x16);
  k_qkvs<<<NN / 16, 256, 0, stream>>>(w.x16, w.WallT, w.ball, w.qkvs16);
  k_node_init<<<(NN * HH) / 256, 256, 0, stream>>>(q16, P.We, P.be, w.qWe, w.qbe,
                                                   w.amaxU, w.denom, w.S1, w.Sw);
  k_zero<<<(NN * DD) / 256, 256, 0, stream>>>(w.out_acc, NN * DD);
  k_alpha<<<(EE * HH) / 256, 256, 0, stream>>>(srcv, dstv, ea, q16, k16, v16,
                                               w.qWe, w.qbe, w.alpha, w.amaxU);
  k_expsum<<<(EE * HH) / 256, 256, 0, stream>>>(dstv, w.amaxU, w.alpha, w.denom);
  k_scatter<<<(EE * HH) / 256, 256, 0, stream>>>(srcv, dstv, ea, w.alpha, w.denom,
                                                 v16, w.S1, w.Sw, w.out_acc);
  k_gate<<<NN / 8, 256, 0, stream>>>(w.out_acc, xr16, w.S1, w.Sw, P.We, P.be,
                                     P.Wbeta, P.bbeta, w.outf16);
  k_lin<<<NN / 16, 64, 0, stream>>>(w.outf16, w.WlinT, P.blin, w.hbuf);
  k_gn_mean<<<2, 256, 0, stream>>>(w.hbuf, w.gmean);
  k_gn_norm<<<2, 256, 0, stream>>>(w.hbuf, w.gmean, P.mean_scale, P.gamma, P.beta_gn, outp);
}

extern "C" void kernel_launch(void* const* d_in, const int* in_sizes, int n_in,
                              void* d_out, int out_size, void* d_ws, size_t ws_size,
                              hipStream_t stream)
{
  (void)in_sizes; (void)n_in; (void)out_size; (void)ws_size;
  const float* x   = (const float*)d_in[0];
  const int*   ei  = (const int*)d_in[1];      // [2,E]
  const float* ea  = (const float*)d_in[2];    // [E,1]
  // d_in[3] = batch_index (unused: batch == node / N_PER by construction)
  BP b0 = get_bp(d_in, 4);
  BP b1 = get_bp(d_in, 21);
  const int* srcv = ei;
  const int* dstv = ei + EE;

  char* p = (char*)d_ws;
  WSBuf w;
  w.qkvs16  = (_Float16*)carve(p, (size_t)4 * NN * DD * 2);   // q|k|v|skip
  w.outf16  = (_Float16*)carve(p, (size_t)NN * DD * 2);
  w.x16     = (_Float16*)carve(p, (size_t)NN * 32 * 2);
  w.WallT   = (_Float16*)carve(p, (size_t)512 * 32 * 2);
  w.WlinT   = (_Float16*)carve(p, (size_t)32 * 128 * 2);
  w.out_acc = (float*)carve(p, (size_t)NN * DD * 4);
  w.alpha   = (float*)carve(p, (size_t)EE * HH * 4);
  w.qWe     = (float*)carve(p, (size_t)NN * HH * 4);
  w.qbe     = (float*)carve(p, (size_t)NN * HH * 4);
  w.denom   = (float*)carve(p, (size_t)NN * HH * 4);
  w.S1      = (float*)carve(p, (size_t)NN * HH * 4);
  w.Sw      = (float*)carve(p, (size_t)NN * HH * 4);
  w.amaxU   = (unsigned*)carve(p, (size_t)NN * HH * 4);
  w.ball    = (float*)carve(p, (size_t)512 * 4);
  w.gmean   = (float*)carve(p, (size_t)NGRAPH * CC * 4);
  w.h0      = (float*)carve(p, (size_t)NN * CC * 4);
  w.hbuf    = (float*)carve(p, (size_t)NN * CC * 4);

  run_block(stream, 4,  x,    b0, w, w.h0,           srcv, dstv, ea);
  run_block(stream, 32, w.h0, b1, w, (float*)d_out,  srcv, dstv, ea);
}
